// feature_31799937859958
// MI455X (gfx1250) — matrix operand order repaired and checked by reading, compiled, not run on hardware
//
#include <hip/hip_runtime.h>

typedef __attribute__((ext_vector_type(16))) _Float16 v16h;
typedef __attribute__((ext_vector_type(8)))  _Float16 v8h;
typedef __attribute__((ext_vector_type(8)))  float    v8f;

#define HW_ (256 * 256)
#define USE_ASYNC_STAGE 1   // 1: global_load_async_to_lds_b128 staging, 0: ld+ds_store

// ---------------- front-end: histogram ----------------
__global__ __launch_bounds__(256) void hist_kernel(const float* __restrict__ x,
                                                   float* __restrict__ counts, int n) {
  int i = blockIdx.x * blockDim.x + threadIdx.x;
  if (i >= n) return;
  int bc = i / HW_;                      // (b*3 + c)
  int bin = (int)floorf(x[i] * 256.f);
  bin = bin < 0 ? 0 : (bin > 255 ? 255 : bin);
  atomicAdd(&counts[bc * 256 + bin], 1.f);
}

// ---------------- front-end: GLCM pair map ----------------
__device__ __forceinline__ int gray8(const float* __restrict__ x, int b, int hw) {
  int r  = (int)floorf(x[(b * 3 + 0) * HW_ + hw] * 255.f);
  int g  = (int)floorf(x[(b * 3 + 1) * HW_ + hw] * 255.f);
  int bb = (int)floorf(x[(b * 3 + 2) * HW_ + hw] * 255.f);
  int l = (19595 * r + 38470 * g + 7471 * bb + 32768) >> 16;
  return l >> 5;
}

__global__ __launch_bounds__(256) void pair_kernel(const float* __restrict__ x,
                                                   unsigned char* __restrict__ pairs, int n) {
  int i = blockIdx.x * blockDim.x + threadIdx.x;
  if (i >= n) return;
  int b = i >> 16, hw = i & 65535, w = hw & 255;
  int g1 = gray8(x, b, hw);
  int g2 = gray8(x, b, (w < 255) ? hw + 1 : hw);
  pairs[i] = (unsigned char)(g1 * 8 + g2);
}

__device__ __forceinline__ int reflect256(int p) {
  return p < 0 ? -p : (p > 255 ? 510 - p : p);
}

// ---------------- front-end: build (B,H,W,4) f16 feature ----------------
__global__ __launch_bounds__(256) void feat_kernel(const float* __restrict__ counts,
                                                   const unsigned char* __restrict__ pairs,
                                                   _Float16* __restrict__ feat, int n) {
  int i = blockIdx.x * blockDim.x + threadIdx.x;
  if (i >= n) return;
  int b = i >> 16, hw = i & 65535, h = hw >> 8, w = hw & 255;
  unsigned char win[25];
  int t = 0;
  for (int dy = -2; dy <= 2; ++dy) {
    int ih = reflect256(h + dy);
    for (int dx = -2; dx <= 2; ++dx) {
      int iw = reflect256(w + dx);
      win[t++] = pairs[(b << 16) + (ih << 8) + iw];
    }
  }
  int cnt = 0;
  for (int a = 0; a < 25; ++a)
    for (int c = 0; c < 25; ++c) cnt += (win[a] == win[c]);
  feat[(size_t)i * 4 + 0] = (_Float16)counts[(b * 3 + 0) * 256 + h];
  feat[(size_t)i * 4 + 1] = (_Float16)counts[(b * 3 + 1) * 256 + h];
  feat[(size_t)i * 4 + 2] = (_Float16)counts[(b * 3 + 2) * 256 + h];
  feat[(size_t)i * 4 + 3] = (_Float16)(float)cnt;
}

// ---- weights (layer 0 generic): OIHW f32 -> KxN f16, K = (ky*3+kx)*Cin+ci ----
__global__ __launch_bounds__(256) void wt_kernel(const float* __restrict__ w,
                                                 _Float16* __restrict__ wT,
                                                 int Cin, int Cout, int n) {
  int i = blockIdx.x * blockDim.x + threadIdx.x;
  if (i >= n) return;
  int co = i / (Cin * 9);
  int ci = (i / 9) % Cin;
  int kk = i % 9;
  int k = kk * Cin + ci;
  wT[(size_t)k * Cout + co] = (_Float16)w[i];
}

// ---- weights (fast layers): B-fragment order, 4KB tile per (kt, nt) ----
__global__ __launch_bounds__(256) void wpack_kernel(const float* __restrict__ w,
                                                    _Float16* __restrict__ wp,
                                                    int Cin, int Cout, int n) {
  int i = blockIdx.x * blockDim.x + threadIdx.x;
  if (i >= n) return;
  int co = i / (Cin * 9);
  int ci = (i / 9) % Cin;
  int kk = i % 9;
  int K = kk * Cin + ci;
  int kt = K >> 5, r = K & 31;
  wp[((size_t)(kt * Cout + co)) * 32 + r] = (_Float16)w[i];
}

// ---------------- generic conv (layer 0, Cin=4): scalar gather ----------------
__global__ __launch_bounds__(256) void conv_wmma_generic(const _Float16* __restrict__ in,
                                                         const _Float16* __restrict__ wT,
                                                         _Float16* __restrict__ out,
                                                         float* __restrict__ stats,
                                                         int Bn, int H, int W, int Cin, int Cout,
                                                         int cinShift, int cinMask) {
  const int lane = threadIdx.x & 31;
  const int wave = blockIdx.x * (blockDim.x >> 5) + (threadIdx.x >> 5);
  const int Mtiles = (Bn * H * W) >> 4;
  const int mt = wave % Mtiles;
  const int nt = wave / Mtiles;

  const int m = lane & 15;
  const int p = mt * 16 + m;
  const int b  = p / (H * W);
  const int hw = p - b * H * W;
  const int oh = hw / W;
  const int ow = hw - oh * W;
  const int kaOff = (lane & 16) ? 8  : 0;
  const int kbOff = (lane & 16) ? 16 : 0;
  const int ncol  = nt * 64 + (lane & 15);

  const int Ktot = 9 * Cin;
  v8f acc0 = {}, acc1 = {}, acc2 = {}, acc3 = {};

  for (int k0 = 0; k0 < Ktot; k0 += 32) {
    v16h a;
#pragma unroll
    for (int j = 0; j < 16; ++j) {
      int K = k0 + kaOff + j + (j & 8);
      _Float16 v = (_Float16)0.f;
      if (K < Ktot) {
        int ci = K & cinMask;
        int t  = K >> cinShift;
        int ky = t >= 6 ? 2 : (t >= 3 ? 1 : 0);
        int kx = t - 3 * ky;
        int ih = oh + ky - 1, iw = ow + kx - 1;
        if (ih >= 0 && ih < H && iw >= 0 && iw < W)
          v = in[((size_t)((b * H + ih) * W + iw)) * Cin + ci];
      }
      a[j] = v;
    }
    v16h b0, b1, b2, b3;
#pragma unroll
    for (int j = 0; j < 16; ++j) {
      int K = k0 + kaOff + j + (j & 8);
      bool ok = K < Ktot;
      size_t base = (size_t)K * Cout + ncol;
      b0[j] = ok ? wT[base]      : (_Float16)0.f;
      b1[j] = ok ? wT[base + 16] : (_Float16)0.f;
      b2[j] = ok ? wT[base + 32] : (_Float16)0.f;
      b3[j] = ok ? wT[base + 48] : (_Float16)0.f;
    }
    acc0 = __builtin_amdgcn_wmma_f32_16x16x32_f16(false, a, false, b0, (short)0, acc0, false, false);
    acc1 = __builtin_amdgcn_wmma_f32_16x16x32_f16(false, a, false, b1, (short)0, acc1, false, false);
    acc2 = __builtin_amdgcn_wmma_f32_16x16x32_f16(false, a, false, b2, (short)0, acc2, false, false);
    acc3 = __builtin_amdgcn_wmma_f32_16x16x32_f16(false, a, false, b3, (short)0, acc3, false, false);
  }

  const int mBase = mt * 16 + ((lane & 16) ? 8 : 0);
  float s0 = 0, q0 = 0, s1 = 0, q1 = 0, s2 = 0, q2 = 0, s3 = 0, q3 = 0;
#pragma unroll
  for (int r = 0; r < 8; ++r) {
    size_t row = (size_t)(mBase + r) * Cout;
    float v;
    v = acc0[r]; out[row + ncol]      = (_Float16)v; s0 += v; q0 += v * v;
    v = acc1[r]; out[row + ncol + 16] = (_Float16)v; s1 += v; q1 += v * v;
    v = acc2[r]; out[row + ncol + 32] = (_Float16)v; s2 += v; q2 += v * v;
    v = acc3[r]; out[row + ncol + 48] = (_Float16)v; s3 += v; q3 += v * v;
  }
  atomicAdd(&stats[ncol],      s0); atomicAdd(&stats[Cout + ncol],      q0);
  atomicAdd(&stats[ncol + 16], s1); atomicAdd(&stats[Cout + ncol + 16], q1);
  atomicAdd(&stats[ncol + 32], s2); atomicAdd(&stats[Cout + ncol + 32], q2);
  atomicAdd(&stats[ncol + 48], s3); atomicAdd(&stats[Cout + ncol + 48], q3);
}

// ---------------- fast conv: vector A loads + (async) LDS-staged B ----------------
__global__ __launch_bounds__(256) void conv_wmma_fast(const _Float16* __restrict__ in,
                                                      const _Float16* __restrict__ wp,
                                                      _Float16* __restrict__ out,
                                                      float* __restrict__ stats,
                                                      int Bn, int H, int W, int Cin, int Cout,
                                                      int cinShift, int cinMask) {
  __shared__ __align__(16) _Float16 lds[2][2048];   // 2 x 4KB weight tiles

  const int tid  = threadIdx.x;
  const int lane = tid & 31;
  const int wave = blockIdx.x * (blockDim.x >> 5) + (tid >> 5);
  const int Mtiles = (Bn * H * W) >> 4;
  const int mt = wave % Mtiles;
  const int nt = wave / Mtiles;                     // uniform across the block

  const int m = lane & 15;
  const int p = mt * 16 + m;
  const int b  = p / (H * W);
  const int hw = p - b * H * W;
  const int oh = hw / W;
  const int ow = hw - oh * W;
  const int kaOff = (lane & 16) ? 16 : 0;
  const int kbOff = (lane & 16) ? 16 : 0;
  const int ncol  = nt * 64 + (lane & 15);

  const int Ktiles = (9 * Cin) >> 5;                // exact (Cin multiple of 32)
  v8f acc0 = {}, acc1 = {}, acc2 = {}, acc3 = {};

  // stage tile 0
#if USE_ASYNC_STAGE
  {
    const _Float16* gsrc = wp + ((size_t)nt * 64) * 32 + tid * 8;
    unsigned ldsoff = (unsigned)(size_t)&lds[0][tid * 8];   // LDS aperture tag is in addr[63:32]
    asm volatile("global_load_async_to_lds_b128 %0, %1, off"
                 :: "v"(ldsoff), "v"(gsrc) : "memory");
    asm volatile("s_wait_asynccnt 0x0" ::: "memory");
  }
#else
  {
    v8h st = *(const v8h*)(wp + ((size_t)nt * 64) * 32 + tid * 8);
    *(v8h*)&lds[0][tid * 8] = st;
  }
#endif
  __syncthreads();

  for (int kt = 0; kt < Ktiles; ++kt) {
    const int pbuf = kt & 1;
    if (kt + 1 < Ktiles) {                          // pre-stage next tile
#if USE_ASYNC_STAGE
      const _Float16* gsrc = wp + ((size_t)(kt + 1) * Cout + nt * 64) * 32 + tid * 8;
      unsigned ldsoff = (unsigned)(size_t)&lds[pbuf ^ 1][tid * 8];
      asm volatile("global_load_async_to_lds_b128 %0, %1, off"
                   :: "v"(ldsoff), "v"(gsrc) : "memory");
#else
      v8h st = *(const v8h*)(wp + ((size_t)(kt + 1) * Cout + nt * 64) * 32 + tid * 8);
      *(v8h*)&lds[pbuf ^ 1][tid * 8] = st;
#endif
    }

    // ---- A fragment: one (ky,kx) tap per K-step; two contiguous b128 loads ----
    const int k0 = kt << 5;
    const int t  = k0 >> cinShift;                  // 0..8
    const int ky = t >= 6 ? 2 : (t >= 3 ? 1 : 0);
    const int kx = t - 3 * ky;
    const int c0 = k0 & cinMask;
    const int ih = oh + ky - 1, iw = ow + kx - 1;
    v16h a = {};
    if (ih >= 0 && ih < H && iw >= 0 && iw < W) {
      const _Float16* ap = in + ((size_t)((b * H + ih) * W + iw)) * Cin + c0 + kaOff;
      v8h lo = *(const v8h*)ap;                     // K = k0+kaOff+0..7
      v8h hi = *(const v8h*)(ap + 8);               // K = k0+kaOff+8..15
      __builtin_prefetch(ap + 32, 0, 0);            // warm next channel group
#pragma unroll
      for (int j = 0; j < 8; ++j) { a[j] = lo[j]; a[j + 8] = hi[j]; }
    }

    // ---- B fragments from LDS: 16 contiguous halfs per fragment ----
    const _Float16* lb = &lds[pbuf][0];
    v16h b0 = *(const v16h*)(lb + ((lane & 15) +  0) * 32 + kbOff);
    v16h b1 = *(const v16h*)(lb + ((lane & 15) + 16) * 32 + kbOff);
    v16h b2 = *(const v16h*)(lb + ((lane & 15) + 32) * 32 + kbOff);
    v16h b3 = *(const v16h*)(lb + ((lane & 15) + 48) * 32 + kbOff);

    acc0 = __builtin_amdgcn_wmma_f32_16x16x32_f16(false, a, false, b0, (short)0, acc0, false, false);
    acc1 = __builtin_amdgcn_wmma_f32_16x16x32_f16(false, a, false, b1, (short)0, acc1, false, false);
    acc2 = __builtin_amdgcn_wmma_f32_16x16x32_f16(false, a, false, b2, (short)0, acc2, false, false);
    acc3 = __builtin_amdgcn_wmma_f32_16x16x32_f16(false, a, false, b3, (short)0, acc3, false, false);

#if USE_ASYNC_STAGE
    asm volatile("s_wait_asynccnt 0x0" ::: "memory");  // own async writes to LDS done
#endif
    __syncthreads();                                   // all waves' staging visible
  }

  // epilogue: store pre-BN f16, accumulate per-channel sum / sumsq
  const int mBase = mt * 16 + ((lane & 16) ? 8 : 0);
  float s0 = 0, q0 = 0, s1 = 0, q1 = 0, s2 = 0, q2 = 0, s3 = 0, q3 = 0;
#pragma unroll
  for (int r = 0; r < 8; ++r) {
    size_t row = (size_t)(mBase + r) * Cout;
    float v;
    v = acc0[r]; out[row + ncol]      = (_Float16)v; s0 += v; q0 += v * v;
    v = acc1[r]; out[row + ncol + 16] = (_Float16)v; s1 += v; q1 += v * v;
    v = acc2[r]; out[row + ncol + 32] = (_Float16)v; s2 += v; q2 += v * v;
    v = acc3[r]; out[row + ncol + 48] = (_Float16)v; s3 += v; q3 += v * v;
  }
  atomicAdd(&stats[ncol],      s0); atomicAdd(&stats[Cout + ncol],      q0);
  atomicAdd(&stats[ncol + 16], s1); atomicAdd(&stats[Cout + ncol + 16], q1);
  atomicAdd(&stats[ncol + 32], s2); atomicAdd(&stats[Cout + ncol + 32], q2);
  atomicAdd(&stats[ncol + 48], s3); atomicAdd(&stats[Cout + ncol + 48], q3);
}

// ---------------- BN + ReLU, vectorized 8 channels/thread ----------------
__global__ __launch_bounds__(256) void bn_relu8(const v8h* __restrict__ in,
                                                v8h* __restrict__ out,
                                                const float* __restrict__ stats,
                                                const float* __restrict__ gamma,
                                                const float* __restrict__ beta,
                                                int n8, int Cout, int cMask, float invCnt) {
  int i = blockIdx.x * blockDim.x + threadIdx.x;
  if (i >= n8) return;
  int c0 = (i * 8) & cMask;
  v8h v = in[i];
  v8h o;
#pragma unroll
  for (int j = 0; j < 8; ++j) {
    int c = c0 + j;
    float mean = stats[c] * invCnt;
    float var  = stats[Cout + c] * invCnt - mean * mean;
    float y = ((float)v[j] - mean) * rsqrtf(var + 1e-5f) * gamma[c] + beta[c];
    o[j] = (_Float16)(y > 0.f ? y : 0.f);
  }
  out[i] = o;
}

// ---------------- maxpool 2x2 NHWC, vectorized 8 channels/thread ----------------
__global__ __launch_bounds__(256) void pool_kernel8(const v8h* __restrict__ in,
                                                    v8h* __restrict__ out,
                                                    int Ho, int Wo, int Cg, int n8) {
  int i = blockIdx.x * blockDim.x + threadIdx.x;
  if (i >= n8) return;
  int cg = i % Cg;
  int t = i / Cg;
  int ow = t % Wo; t /= Wo;
  int oh = t % Ho;
  int b  = t / Ho;
  int W = Wo * 2;
  size_t base = ((size_t)((b * (Ho * 2) + 2 * oh) * W + 2 * ow)) * Cg + cg;
  v8h a0 = in[base],                a1 = in[base + Cg];
  v8h a2 = in[base + (size_t)W * Cg], a3 = in[base + (size_t)W * Cg + Cg];
  v8h o;
#pragma unroll
  for (int j = 0; j < 8; ++j) {
    float mx = fmaxf(fmaxf((float)a0[j], (float)a1[j]), fmaxf((float)a2[j], (float)a3[j]));
    o[j] = (_Float16)mx;
  }
  out[i] = o;
}

// ---------------- final maxpool: NHWC f16 -> NCHW f32 into d_out ----------------
__global__ __launch_bounds__(256) void pool_final(const _Float16* __restrict__ in,
                                                  float* __restrict__ out,
                                                  int Ho, int Wo, int C, int cMask, int n) {
  int i = blockIdx.x * blockDim.x + threadIdx.x;
  if (i >= n) return;
  int c = i & cMask;
  int t = i / C;
  int ow = t % Wo; t /= Wo;
  int oh = t % Ho;
  int b  = t / Ho;
  int W = Wo * 2;
  size_t base = ((size_t)((b * (Ho * 2) + 2 * oh) * W + 2 * ow)) * C + c;
  float a0 = (float)in[base],                 a1 = (float)in[base + C];
  float a2 = (float)in[base + (size_t)W * C], a3 = (float)in[base + (size_t)W * C + C];
  out[(((size_t)b * C + c) * Ho + oh) * Wo + ow] = fmaxf(fmaxf(a0, a1), fmaxf(a2, a3));
}

// ---------------- driver ----------------
extern "C" void kernel_launch(void* const* d_in, const int* in_sizes, int n_in,
                              void* d_out, int out_size, void* d_ws, size_t ws_size,
                              hipStream_t stream) {
  (void)in_sizes; (void)n_in; (void)out_size; (void)ws_size;
  const float* x = (const float*)d_in[0];
  char* ws = (char*)d_ws;

  const size_t S = (size_t)16 * 256 * 256 * 64 * sizeof(_Float16);  // 134 MB act buffer
  _Float16* buf0 = (_Float16*)ws;
  _Float16* buf1 = (_Float16*)(ws + S);
  float* stats          = (float*)(ws + 2 * S);                      // 1024 f32
  float* hist           = (float*)(ws + 2 * S + 4096);               // 16*3*256 f32
  unsigned char* pairs  = (unsigned char*)(ws + 2 * S + 4096 + 49152);
  _Float16* wTbase      = (_Float16*)(ws + 2 * S + 4096 + 49152 + (1 << 20));

  const int Cin_[8]  = {4, 64, 64, 128, 128, 256, 256, 512};
  const int Cout_[8] = {64, 64, 128, 128, 256, 256, 512, 512};
  const int Hh_[8]   = {256, 256, 128, 128, 64, 64, 32, 32};

  _Float16* wTs[8];
  size_t off = 0;
  for (int L = 0; L < 8; ++L) {
    wTs[L] = wTbase + off;
    int nw = Cout_[L] * Cin_[L] * 9;
    if (L == 0)
      wt_kernel<<<(nw + 255) / 256, 256, 0, stream>>>((const float*)d_in[1 + 3 * L],
                                                      wTs[L], Cin_[L], Cout_[L], nw);
    else
      wpack_kernel<<<(nw + 255) / 256, 256, 0, stream>>>((const float*)d_in[1 + 3 * L],
                                                         wTs[L], Cin_[L], Cout_[L], nw);
    off += (size_t)nw;
  }

  // front-end features -> buf0 (B,256,256,4) f16
  hipMemsetAsync(hist, 0, 12288 * sizeof(float), stream);
  int nx = 16 * 3 * HW_;
  hist_kernel<<<(nx + 255) / 256, 256, 0, stream>>>(x, hist, nx);
  int np = 16 * HW_;
  pair_kernel<<<(np + 255) / 256, 256, 0, stream>>>(x, pairs, np);
  feat_kernel<<<(np + 255) / 256, 256, 0, stream>>>(hist, pairs, buf0, np);

  // VGG8
  _Float16* cur = buf0;
  _Float16* other = buf1;
  for (int L = 0; L < 8; ++L) {
    int Cin = Cin_[L], Cout = Cout_[L], H = Hh_[L];
    hipMemsetAsync(stats, 0, 1024 * sizeof(float), stream);
    int Mtiles = (16 * H * H) >> 4;
    int waves = Mtiles * (Cout >> 6);               // multiple of 8: grid is exact
    if (L == 0)
      conv_wmma_generic<<<waves / 8, 256, 0, stream>>>(cur, wTs[L], other, stats,
                                                       16, H, H, Cin, Cout,
                                                       __builtin_ctz((unsigned)Cin), Cin - 1);
    else
      conv_wmma_fast<<<waves / 8, 256, 0, stream>>>(cur, wTs[L], other, stats,
                                                    16, H, H, Cin, Cout,
                                                    __builtin_ctz((unsigned)Cin), Cin - 1);
    int n8 = (16 * H * H * Cout) / 8;
    bn_relu8<<<(n8 + 255) / 256, 256, 0, stream>>>((const v8h*)other, (v8h*)cur, stats,
                                                   (const float*)d_in[2 + 3 * L],
                                                   (const float*)d_in[3 + 3 * L],
                                                   n8, Cout, Cout - 1, 1.f / (16.f * H * H));
    if (L & 1) {                                    // pool after 12, 22, 32, 42
      int Ho = H / 2;
      if (L == 7) {
        int npo = 16 * Ho * Ho * Cout;
        pool_final<<<(npo + 255) / 256, 256, 0, stream>>>(cur, (float*)d_out,
                                                          Ho, Ho, Cout, Cout - 1, npo);
      } else {
        int npo8 = (16 * Ho * Ho * Cout) / 8;
        pool_kernel8<<<(npo8 + 255) / 256, 256, 0, stream>>>((const v8h*)cur, (v8h*)other,
                                                             Ho, Ho, Cout / 8, npo8);
        _Float16* tmp = cur; cur = other; other = tmp;
      }
    }
  }
}
